// OpenPanguMoE_87711822119115
// MI455X (gfx1250) — matrix operand order repaired and checked by reading, not compiled, not run on hardware
//
#include <hip/hip_runtime.h>

#define T_TOKENS 4096
#define H_DIM    2048
#define E_EXPERTS 64
#define TOPK     8
#define I_DIM    512
#define N1       1024          // 2*I
#define N2       2048          // H
#define PAIRS    (T_TOKENS * TOPK)       // 32768
#define MROWS    (PAIRS + T_TOKENS)      // 36864 (routed pairs + shared-expert rows)
#define TM       128
#define TN       128
#define KB       64            // two WMMA k-slices per LDS stage
#define LDSW     (KB + 4)      // 68 ushorts row stride (even, 2-dword pad)
#define MAX_TILES 352
#define SCALING  2.5f

typedef __attribute__((ext_vector_type(16))) __bf16 v16bf;
typedef __attribute__((ext_vector_type(8)))  float  v8f;
typedef __attribute__((ext_vector_type(4)))  unsigned v4u;
typedef __attribute__((ext_vector_type(8)))  int    v8i;
typedef __attribute__((ext_vector_type(4)))  int    v4i;

union BF16x16 { v16bf v; unsigned u[8]; };
union F32x32  { float4 v4[8]; float f[32]; };

__device__ __forceinline__ unsigned short f2bf(float f) {
    union { float f; unsigned u; } v; v.f = f;
    unsigned u = v.u + 0x7FFFu + ((v.u >> 16) & 1u);   // RNE truncate
    return (unsigned short)(u >> 16);
}
__device__ __forceinline__ float bf2f(unsigned short b) {
    union { unsigned u; float f; } v; v.u = ((unsigned)b) << 16;
    return v.f;
}
__device__ __forceinline__ unsigned pack2(unsigned short lo, unsigned short hi) {
    return (unsigned)lo | ((unsigned)hi << 16);
}

// Issue one TDM 2D-tile load: [tile_dim1=128 rows x tile_dim0=64 bf16] from a
// row-major bf16 tensor (row stride 512 elems) into LDS with a 2-dword pad per
// 32-dword row (=> 68-ushort LDS row stride). tensor_dim1=rowCnt gives OOB
// zero-fill for partial tiles.
__device__ __forceinline__ void tdm_load_A(unsigned lds_off,
                                           const unsigned short* gptr,
                                           int rowCnt) {
    unsigned long long ga = (unsigned long long)(size_t)gptr;
    v4u g0;
    g0[0] = 1u;                                          // count=1, no gather
    g0[1] = lds_off;                                     // lds_addr
    g0[2] = (unsigned)(ga & 0xFFFFFFFFu);                // global_addr[31:0]
    g0[3] = (unsigned)((ga >> 32) & 0x01FFFFFFu)         // global_addr[56:32]
            | (2u << 30);                                // type=2 ("image")
    v8i g1;
    g1[0] = (1 << 16)        // data_size = 2 bytes
          | (1 << 20)        // pad_enable
          | (4 << 22)        // pad_interval: 32 dwords
          | (1 << 25);       // pad_amount:   2 dwords
    g1[1] = (KB << 16);                     // abar_addr=0 | tensor_dim0 lo16 = 64
    g1[2] = ((rowCnt & 0xFFFF) << 16);      // tensor_dim0 hi16=0 | tensor_dim1 lo16
    g1[3] = (KB << 16);                     // tensor_dim1 hi16=0 | tile_dim0 = 64
    g1[4] = TM;                             // tile_dim1 = 128 | tile_dim2 = 0
    g1[5] = I_DIM;                          // tensor_dim0_stride lo32 = 512
    g1[6] = 0;
    g1[7] = 0;
    v4i z4 = {0, 0, 0, 0};
    v8i z8 = {0, 0, 0, 0, 0, 0, 0, 0};
    __builtin_amdgcn_tensor_load_to_lds(g0, g1, z4, z4, z8, 0);
}

// ---------------- router: logits + sigmoid + top-8 + renormalize ------------
__global__ void router_kernel(const float* __restrict__ hidden,
                              const float* __restrict__ gate_w,
                              int* __restrict__ topk_ids,
                              float* __restrict__ topk_w) {
    __shared__ float sc[E_EXPERTS];
    const int t = blockIdx.x;
    const int e = threadIdx.x;
    const float* hrow = hidden + (size_t)t * H_DIM;
    float acc = 0.f;
    #pragma unroll 4
    for (int k = 0; k < H_DIM; ++k)
        acc += hrow[k] * gate_w[(size_t)k * E_EXPERTS + e];
    sc[e] = 1.f / (1.f + __expf(-acc));
    __syncthreads();
    if (e == 0) {
        int   ids[TOPK];
        float ws_[TOPK];
        float wsum = 0.f;
        for (int k = 0; k < TOPK; ++k) {
            float best = -1.f; int bi = 0;
            for (int j = 0; j < E_EXPERTS; ++j)
                if (sc[j] > best) { best = sc[j]; bi = j; }
            ids[k] = bi; ws_[k] = best; wsum += best; sc[bi] = -2.f;
        }
        float inv = 1.f / wsum;
        for (int k = 0; k < TOPK; ++k) {
            topk_ids[t * TOPK + k] = ids[k];
            topk_w[t * TOPK + k]   = ws_[k] * inv;
        }
    }
}

// meta layout: [0:64) counts, [64:128) cursors, [128:192) offsets, [192] numTiles
__global__ void init_kernel(int* __restrict__ meta) {
    meta[threadIdx.x] = 0;
}

__global__ void hist_kernel(const int* __restrict__ ids, int* __restrict__ meta) {
    int p = blockIdx.x * blockDim.x + threadIdx.x;
    if (p < PAIRS) atomicAdd(&meta[ids[p]], 1);
}

__global__ void plan_kernel(int* __restrict__ meta, int4* __restrict__ tiles) {
    if (threadIdx.x != 0) return;
    int* counts  = meta;
    int* offsets = meta + 128;
    int off = 0;
    for (int e = 0; e < E_EXPERTS; ++e) { offsets[e] = off; off += counts[e]; }
    int nt = 0;
    for (int e = 0; e <= E_EXPERTS; ++e) {          // e==E_EXPERTS: shared expert
        int cnt = (e < E_EXPERTS) ? counts[e] : T_TOKENS;
        int rs  = (e < E_EXPERTS) ? offsets[e] : PAIRS;
        for (int m0 = 0; m0 < cnt; m0 += TM) {
            int c = cnt - m0; if (c > TM) c = TM;
            tiles[nt++] = make_int4(e, rs + m0, c, 0);
        }
    }
    meta[192] = nt;
    for (int i = nt; i < MAX_TILES; ++i) tiles[i] = make_int4(0, 0, 0, 0);
}

__global__ void scatter_kernel(const int* __restrict__ ids,
                               const float* __restrict__ tw,
                               int* __restrict__ meta,
                               int* __restrict__ rowOf,
                               float* __restrict__ wgt,
                               int* __restrict__ invpos) {
    int p = blockIdx.x * blockDim.x + threadIdx.x;
    if (p >= MROWS) return;
    if (p < PAIRS) {
        int e   = ids[p];
        int pos = meta[128 + e] + atomicAdd(&meta[64 + e], 1);
        rowOf[pos]  = p / TOPK;
        wgt[pos]    = tw[p];
        invpos[p]   = pos;
    } else {
        int t = p - PAIRS;               // shared-expert rows: identity mapping
        rowOf[PAIRS + t] = t;
        wgt[PAIRS + t]   = 1.f;
    }
}

// ---------------- grouped GEMM1: gu = gather(hidden) @ w13[e]  (bf16 WMMA) ---
__global__ __launch_bounds__(256)
void moe_gemm1_kernel(const float* __restrict__ hidden,
                      const float* __restrict__ w13,
                      const float* __restrict__ shared_w13,
                      const int* __restrict__ rowOf,
                      const int* __restrict__ meta,
                      const int4* __restrict__ tiles,
                      unsigned short* __restrict__ gu) {
    if ((int)blockIdx.x >= meta[192]) return;
    const int4 td = tiles[blockIdx.x];
    const int expert = td.x, rowStart = td.y, rowCnt = td.z;
    const int n0 = blockIdx.y * TN;
    const float* B = (expert < E_EXPERTS) ? (w13 + (size_t)expert * H_DIM * N1)
                                          : shared_w13;

    __shared__ unsigned short As[TM][LDSW];
    __shared__ unsigned short BsT[TN][LDSW];    // transposed: [n][k]

    const int tid  = threadIdx.x;
    const int lane = tid & 31;
    const int wave = tid >> 5;
    const int mBase = (wave & 3) * 32;          // 4 waves along M
    const int nBase = (wave >> 2) * 64;         // 2 waves along N

    // A-tile loading: 128 rows x 64 k, 32 floats/thread
    const int aRow = tid >> 1;
    const int aCol = (tid & 1) * 32;
    const bool aValid = aRow < rowCnt;
    const int gRow = aValid ? rowOf[rowStart + aRow] : 0;
    const float* aBase = hidden + (size_t)gRow * H_DIM + aCol;

    // B-tile loading: 64 k x 128 n, 32 floats/thread
    const int bRow = tid >> 2;
    const int bCol = (tid & 3) * 32;

    v8f acc[2][4] = {};

    for (int k0 = 0; k0 < H_DIM; k0 += KB) {
        F32x32 aReg, bReg;
        if (aValid) {
            const float4* p = (const float4*)(aBase + k0);
            #pragma unroll
            for (int i = 0; i < 8; ++i) aReg.v4[i] = p[i];
        } else {
            #pragma unroll
            for (int i = 0; i < 32; ++i) aReg.f[i] = 0.f;
        }
        {
            const float4* p = (const float4*)(B + (size_t)(k0 + bRow) * N1 + n0 + bCol);
            #pragma unroll
            for (int i = 0; i < 8; ++i) bReg.v4[i] = p[i];
        }
        if (k0 + KB < H_DIM)   // prefetch next B tile stream into caches
            __builtin_prefetch(B + (size_t)(k0 + KB + bRow) * N1 + n0 + bCol, 0, 3);
        __syncthreads();
        {
            unsigned* ar = (unsigned*)&As[aRow][aCol];
            #pragma unroll
            for (int p = 0; p < 16; ++p)
                ar[p] = pack2(f2bf(aReg.f[2 * p]), f2bf(aReg.f[2 * p + 1]));
            #pragma unroll
            for (int i = 0; i < 32; ++i)
                BsT[bCol + i][bRow] = f2bf(bReg.f[i]);
        }
        __syncthreads();

        #pragma unroll
        for (int ks = 0; ks < KB; ks += 32) {
            const int arow0 = mBase + (lane & 15);
            const int kgA   = ks + (lane >> 4) * 8;
            BF16x16 af[2];
            #pragma unroll
            for (int i = 0; i < 2; ++i) {
                #pragma unroll
                for (int p = 0; p < 4; ++p) {
                    af[i].u[p]     = *(const unsigned*)&As[arow0 + i * 16][kgA + 2 * p];
                    af[i].u[p + 4] = *(const unsigned*)&As[arow0 + i * 16][kgA + 16 + 2 * p];
                }
            }
            const int bcol0 = nBase + (lane & 15);
            const int kgB   = ks + (lane >> 4) * 8;
            BF16x16 bfr[4];
            #pragma unroll
            for (int j = 0; j < 4; ++j) {
                #pragma unroll
                for (int p = 0; p < 4; ++p) {
                    bfr[j].u[p]     = *(const unsigned*)&BsT[bcol0 + j * 16][kgB + 2 * p];
                    bfr[j].u[p + 4] = *(const unsigned*)&BsT[bcol0 + j * 16][kgB + 16 + 2 * p];
                }
            }
            #pragma unroll
            for (int i = 0; i < 2; ++i)
                #pragma unroll
                for (int j = 0; j < 4; ++j)
                    acc[i][j] = __builtin_amdgcn_wmma_f32_16x16x32_bf16(
                        false, af[i].v, false, bfr[j].v, (short)0, acc[i][j],
                        false, false);
        }
    }

    const int mr = (lane >> 4) * 8;
    const int nc = lane & 15;
    #pragma unroll
    for (int i = 0; i < 2; ++i)
        #pragma unroll
        for (int j = 0; j < 4; ++j)
            #pragma unroll
            for (int r = 0; r < 8; ++r) {
                int m = mBase + i * 16 + mr + r;
                if (m < rowCnt) {
                    int col = n0 + nBase + j * 16 + nc;
                    gu[(size_t)(rowStart + m) * N1 + col] = f2bf(acc[i][j][r]);
                }
            }
}

// ---------------- SiLU(gate) * up ------------------------------------------
__global__ void silu_kernel(const unsigned short* __restrict__ gu,
                            unsigned short* __restrict__ h) {
    int idx = blockIdx.x * blockDim.x + threadIdx.x;
    if (idx >= MROWS * I_DIM) return;
    int r = idx >> 9;
    int i = idx & (I_DIM - 1);
    float g = bf2f(gu[(size_t)r * N1 + i]);
    float u = bf2f(gu[(size_t)r * N1 + I_DIM + i]);
    float s = g / (1.f + __expf(-g));
    h[idx] = f2bf(s * u);
}

// ---------- grouped GEMM2: out = (h @ w2[e]) * wgt  (TDM A, bf16 WMMA) ------
__global__ __launch_bounds__(256)
void moe_gemm2_kernel(const unsigned short* __restrict__ h,
                      const float* __restrict__ w2,
                      const float* __restrict__ shared_w2,
                      const float* __restrict__ wgt,
                      const int* __restrict__ meta,
                      const int4* __restrict__ tiles,
                      float* __restrict__ out_sorted) {
    if ((int)blockIdx.x >= meta[192]) return;
    const int4 td = tiles[blockIdx.x];
    const int expert = td.x, rowStart = td.y, rowCnt = td.z;
    const int n0 = blockIdx.y * TN;
    const float* B = (expert < E_EXPERTS) ? (w2 + (size_t)expert * I_DIM * N2)
                                          : shared_w2;

    __shared__ unsigned short As[2][TM][LDSW];   // double-buffered TDM dest
    __shared__ unsigned short BsT[TN][LDSW];

    const int tid  = threadIdx.x;
    const int lane = tid & 31;
    const int wave = tid >> 5;
    const int mBase = (wave & 3) * 32;
    const int nBase = (wave >> 2) * 64;

    const int bRow = tid >> 2;                   // 64 k-rows, 4 threads/row
    const int bCol = (tid & 3) * 32;

    const unsigned ldsA0 = (unsigned)(size_t)&As[0][0][0];
    const unsigned ldsA1 = (unsigned)(size_t)&As[1][0][0];
    const unsigned short* aTile = h + (size_t)rowStart * I_DIM;

    constexpr int STAGES = I_DIM / KB;           // 8

    if (wave == 0)                               // prologue: stage 0 DMA
        tdm_load_A(ldsA0, aTile, rowCnt);

    v8f acc[2][4] = {};

    for (int s = 0; s < STAGES; ++s) {
        const int k0 = s * KB;
        F32x32 bReg;
        {
            const float4* p = (const float4*)(B + (size_t)(k0 + bRow) * N2 + n0 + bCol);
            #pragma unroll
            for (int i = 0; i < 8; ++i) bReg.v4[i] = p[i];
        }
        if (k0 + KB < I_DIM)
            __builtin_prefetch(B + (size_t)(k0 + KB + bRow) * N2 + n0 + bCol, 0, 3);
        __syncthreads();                         // everyone done with prev stage LDS
        if (wave == 0 && s + 1 < STAGES)         // kick next-stage DMA (other buffer)
            tdm_load_A(((s + 1) & 1) ? ldsA1 : ldsA0, aTile + (k0 + KB), rowCnt);
        {
            #pragma unroll
            for (int i = 0; i < 32; ++i)
                BsT[bCol + i][bRow] = f2bf(bReg.f[i]);
        }
        if (wave == 0) {                         // stage-s tile must have landed
            if (s + 1 < STAGES) __builtin_amdgcn_s_wait_tensorcnt(1);
            else                __builtin_amdgcn_s_wait_tensorcnt(0);
        }
        __syncthreads();

        const int buf = s & 1;
        #pragma unroll
        for (int ks = 0; ks < KB; ks += 32) {
            const int arow0 = mBase + (lane & 15);
            const int kgA   = ks + (lane >> 4) * 8;
            BF16x16 af[2];
            #pragma unroll
            for (int i = 0; i < 2; ++i) {
                #pragma unroll
                for (int p = 0; p < 4; ++p) {
                    af[i].u[p]     = *(const unsigned*)&As[buf][arow0 + i * 16][kgA + 2 * p];
                    af[i].u[p + 4] = *(const unsigned*)&As[buf][arow0 + i * 16][kgA + 16 + 2 * p];
                }
            }
            const int bcol0 = nBase + (lane & 15);
            const int kgB   = ks + (lane >> 4) * 8;
            BF16x16 bfr[4];
            #pragma unroll
            for (int j = 0; j < 4; ++j) {
                #pragma unroll
                for (int p = 0; p < 4; ++p) {
                    bfr[j].u[p]     = *(const unsigned*)&BsT[bcol0 + j * 16][kgB + 2 * p];
                    bfr[j].u[p + 4] = *(const unsigned*)&BsT[bcol0 + j * 16][kgB + 16 + 2 * p];
                }
            }
            #pragma unroll
            for (int i = 0; i < 2; ++i)
                #pragma unroll
                for (int j = 0; j < 4; ++j)
                    acc[i][j] = __builtin_amdgcn_wmma_f32_16x16x32_bf16(
                        false, af[i].v, false, bfr[j].v, (short)0, acc[i][j],
                        false, false);
        }
    }

    const int mr = (lane >> 4) * 8;
    const int nc = lane & 15;
    #pragma unroll
    for (int i = 0; i < 2; ++i)
        #pragma unroll
        for (int j = 0; j < 4; ++j)
            #pragma unroll
            for (int r = 0; r < 8; ++r) {
                int m = mBase + i * 16 + mr + r;
                if (m < rowCnt) {
                    int row = rowStart + m;
                    int col = n0 + nBase + j * 16 + nc;
                    out_sorted[(size_t)row * N2 + col] = acc[i][j][r] * wgt[row];
                }
            }
}

// ---------------- combine: 2.5 * sum_k routed + shared ----------------------
__global__ void combine_kernel(const float* __restrict__ out_sorted,
                               const int* __restrict__ invpos,
                               float* __restrict__ out) {
    __shared__ int pos[TOPK];
    const int t = blockIdx.x;
    if (threadIdx.x < TOPK) pos[threadIdx.x] = invpos[t * TOPK + threadIdx.x];
    __syncthreads();
    const int c0 = threadIdx.x * 8;
    float acc[8];
    #pragma unroll
    for (int i = 0; i < 8; ++i) acc[i] = 0.f;
    for (int k = 0; k < TOPK; ++k) {
        const float* rp = out_sorted + (size_t)pos[k] * N2 + c0;
        #pragma unroll
        for (int i = 0; i < 8; ++i) acc[i] += rp[i];
    }
    const float* sp = out_sorted + (size_t)(PAIRS + t) * N2 + c0;
    #pragma unroll
    for (int i = 0; i < 8; ++i)
        out[(size_t)t * N2 + c0 + i] = SCALING * acc[i] + sp[i];
}

extern "C" void kernel_launch(void* const* d_in, const int* in_sizes, int n_in,
                              void* d_out, int out_size, void* d_ws, size_t ws_size,
                              hipStream_t stream) {
    (void)in_sizes; (void)n_in; (void)out_size; (void)ws_size;
    const float* hidden      = (const float*)d_in[0];
    const float* gate_w      = (const float*)d_in[1];
    const float* w13         = (const float*)d_in[2];
    const float* w2          = (const float*)d_in[3];
    const float* shared_w13  = (const float*)d_in[4];
    const float* shared_w2   = (const float*)d_in[5];
    float* out = (float*)d_out;

    char* ws = (char*)d_ws;
    size_t off = 0;
    auto alloc = [&](size_t bytes) -> void* {
        void* p = ws + off;
        off = (off + bytes + 255) & ~(size_t)255;
        return p;
    };
    int*   topk_ids   = (int*)  alloc((size_t)PAIRS * 4);
    float* topk_w     = (float*)alloc((size_t)PAIRS * 4);
    int*   meta       = (int*)  alloc(256 * 4);
    int4*  tiles      = (int4*) alloc((size_t)MAX_TILES * 16);
    int*   rowOf      = (int*)  alloc((size_t)MROWS * 4);
    float* wgt        = (float*)alloc((size_t)MROWS * 4);
    int*   invpos     = (int*)  alloc((size_t)PAIRS * 4);
    unsigned short* gu   = (unsigned short*)alloc((size_t)MROWS * N1 * 2);
    unsigned short* hbuf = (unsigned short*)alloc((size_t)MROWS * I_DIM * 2);
    float* out_sorted = (float*)alloc((size_t)MROWS * N2 * 4);

    init_kernel<<<1, 256, 0, stream>>>(meta);
    router_kernel<<<T_TOKENS, E_EXPERTS, 0, stream>>>(hidden, gate_w, topk_ids, topk_w);
    hist_kernel<<<(PAIRS + 255) / 256, 256, 0, stream>>>(topk_ids, meta);
    plan_kernel<<<1, 32, 0, stream>>>(meta, tiles);
    scatter_kernel<<<(MROWS + 255) / 256, 256, 0, stream>>>(topk_ids, topk_w, meta,
                                                            rowOf, wgt, invpos);
    moe_gemm1_kernel<<<dim3(MAX_TILES, N1 / TN), 256, 0, stream>>>(
        hidden, w13, shared_w13, rowOf, meta, tiles, gu);
    silu_kernel<<<(MROWS * I_DIM + 255) / 256, 256, 0, stream>>>(gu, hbuf);
    moe_gemm2_kernel<<<dim3(MAX_TILES, N2 / TN), 256, 0, stream>>>(
        hbuf, w2, shared_w2, wgt, meta, tiles, out_sorted);
    combine_kernel<<<T_TOKENS, 256, 0, stream>>>(out_sorted, invpos, out);
}
